// CrossAttentionLayer_13829794693724
// MI455X (gfx1250) — matrix operand order repaired and checked by reading, not compiled, not run on hardware
//
#include <hip/hip_runtime.h>

typedef __attribute__((ext_vector_type(16))) _Float16 v16h;
typedef __attribute__((ext_vector_type(8)))  _Float16 v8h;
typedef __attribute__((ext_vector_type(8)))  float    v8f;
typedef __attribute__((ext_vector_type(4)))  int      v4i;

#define NB   8
#define SEQ  4096
#define CH   64
#define BLR  (NB * SEQ)          // total rows = 32768
#define LOG2E 1.4426950408889634f

// ---------------- CDNA5 async global->LDS path ---------------------------
#if defined(__gfx1250__) && __has_builtin(__builtin_amdgcn_global_load_async_to_lds_b128)
#define ASYNC_LDS 1
#else
#define ASYNC_LDS 0
#endif

#if ASYNC_LDS
typedef __attribute__((address_space(1))) v4i glob_v4i;
typedef __attribute__((address_space(3))) v4i lds_v4i;

__device__ __forceinline__ void async_b128(void* lds_dst, const void* gsrc) {
    __builtin_amdgcn_global_load_async_to_lds_b128(
        (glob_v4i*)(uintptr_t)gsrc,
        (lds_v4i*)(uint32_t)(uintptr_t)lds_dst,
        0, 0);
}
#if __has_builtin(__builtin_amdgcn_s_wait_asynccnt)
#define WAIT_ASYNC0()                            \
    do {                                         \
        __builtin_amdgcn_s_wait_asynccnt(0);     \
        asm volatile("" ::: "memory");           \
    } while (0)
#else
#define WAIT_ASYNC0() asm volatile("s_wait_asynccnt 0x0" ::: "memory")
#endif
#else
#define WAIT_ASYNC0() do {} while (0)
#endif

// ---------------- helpers -----------------------------------------------

__device__ __forceinline__ v16h join8(v8h lo, v8h hi) {
    v16h r;
#pragma unroll
    for (int i = 0; i < 8; ++i) { r[i] = lo[i]; r[8 + i] = hi[i]; }
    return r;
}

// A-fragment (16x32 f16) from an f32 row-major row; K-step h, lane group g.
__device__ __forceinline__ v16h afrag_from_f32(const float* __restrict__ row, int h, int g) {
    const float* p0 = row + 32 * h + 16 * g;
    const float* p1 = row + 32 * h + 16 * g + 8;
    v8h lo, hi;
#pragma unroll
    for (int i = 0; i < 8; ++i) { lo[i] = (_Float16)p0[i]; hi[i] = (_Float16)p1[i]; }
    return join8(lo, hi);
}

// A-fragment from an f16 row-major row; off = 32*h + 16*g
__device__ __forceinline__ v16h afrag_from_f16(const _Float16* __restrict__ row, int off) {
    v8h lo = *(const v8h*)(row + off);
    v8h hi = *(const v8h*)(row + off + 8);
    return join8(lo, hi);
}

#define WMMA_F16(A, Bf, C) \
    __builtin_amdgcn_wmma_f32_16x16x32_f16(false, (A), false, (Bf), (short)0, (C), false, false)

// ---------------- weight prep -------------------------------------------
__global__ void prep_weights(const float* __restrict__ wq, const float* __restrict__ wk,
                             const float* __restrict__ wv, const float* __restrict__ wd,
                             const float* __restrict__ wo,
                             _Float16* __restrict__ wqT, _Float16* __restrict__ wkT,
                             _Float16* __restrict__ wvT, _Float16* __restrict__ wdT,
                             _Float16* __restrict__ woT) {
    for (int i = threadIdx.x; i < 64 * 64; i += blockDim.x) {
        int c = i >> 6, j = i & 63;
        wqT[c * 64 + j] = (_Float16)wq[j * 64 + c];
        wkT[c * 64 + j] = (_Float16)wk[j * 64 + c];
        wvT[c * 64 + j] = (_Float16)wv[j * 64 + c];
        woT[c * 64 + j] = (_Float16)wo[j * 64 + c];
    }
    for (int i = threadIdx.x; i < 64 * 192; i += blockDim.x) {
        int c = i / 192, kk = i % 192;
        wdT[c * 192 + kk] = (_Float16)wd[kk * 64 + c];
    }
}

// ---------------- 1x1 conv (pointwise GEMM): Y = X @ W + b ---------------
template <bool HALF_OUT>
__global__ __launch_bounds__(256) void pw_gemm(const float* __restrict__ X,
                                               const _Float16* __restrict__ wT_g,
                                               const float* __restrict__ bias,
                                               void* __restrict__ Y) {
    __shared__ __align__(32) _Float16 wT[64 * 64];
    for (int i = threadIdx.x; i < (64 * 64) / 8; i += blockDim.x)
        *(v8h*)&wT[i * 8] = *(const v8h*)&wT_g[i * 8];
    __syncthreads();

    const int w = threadIdx.x >> 5, lane = threadIdx.x & 31;
    const int g = lane >> 4, Lm = lane & 15;
    const int r0 = blockIdx.x * 128 + 16 * w;

    const float* row = X + (size_t)(r0 + Lm) * 64;
    v16h a0 = afrag_from_f32(row, 0, g);
    v16h a1 = afrag_from_f32(row, 1, g);

    v8f acc[4];
#pragma unroll
    for (int nt = 0; nt < 4; ++nt) {
        float bv = bias[16 * nt + Lm];
        v8f c;
#pragma unroll
        for (int i = 0; i < 8; ++i) c[i] = bv;
        v16h b0 = *(const v16h*)&wT[(16 * nt + Lm) * 64 + 16 * g];
        v16h b1 = *(const v16h*)&wT[(16 * nt + Lm) * 64 + 32 + 16 * g];
        c = WMMA_F16(a0, b0, c);
        c = WMMA_F16(a1, b1, c);
        acc[nt] = c;
    }

#pragma unroll
    for (int nt = 0; nt < 4; ++nt)
#pragma unroll
        for (int r = 0; r < 8; ++r) {
            size_t row_o = (size_t)(r0 + r + 8 * g);
            int col = 16 * nt + Lm;
            if constexpr (HALF_OUT)
                ((_Float16*)Y)[row_o * 64 + col] = (_Float16)acc[nt][r];
            else
                ((float*)Y)[row_o * 64 + col] = acc[nt][r];
        }
}

// ---------------- K=3 conv as K=192 GEMM ---------------------------------
// oscale folds softmax's log2(e) into Q_ at zero marginal cost.
__global__ __launch_bounds__(256) void conv3_proj(const _Float16* __restrict__ tmp,
                                                  const _Float16* __restrict__ wdT_g,
                                                  const float* __restrict__ bias,
                                                  _Float16* __restrict__ Yrow,
                                                  _Float16* __restrict__ Ytr,
                                                  int transposed, float oscale) {
    __shared__ __align__(32) _Float16 wdT[64 * 192];
    for (int i = threadIdx.x; i < (64 * 192) / 8; i += blockDim.x)
        *(v8h*)&wdT[i * 8] = *(const v8h*)&wdT_g[i * 8];
    __syncthreads();

    const int w = threadIdx.x >> 5, lane = threadIdx.x & 31;
    const int g = lane >> 4, Lm = lane & 15;
    const int r0 = blockIdx.x * 128 + 16 * w;
    const int G = r0 + Lm;
    const int b = G >> 12, l = G & 4095;

    v8f acc[4];
#pragma unroll
    for (int nt = 0; nt < 4; ++nt) {
        float bv = bias[16 * nt + Lm];
#pragma unroll
        for (int i = 0; i < 8; ++i) acc[nt][i] = bv;
    }

#pragma unroll
    for (int k = 0; k < 3; ++k) {
        int lp = l + k - 1;
        bool valid = (lp >= 0) && (lp < SEQ);
        const _Float16* row = tmp + ((size_t)(b << 12) + (valid ? lp : 0)) * 64;
        _Float16 sc = valid ? (_Float16)1.0f : (_Float16)0.0f;
#pragma unroll
        for (int h = 0; h < 2; ++h) {
            v16h a = afrag_from_f16(row, 32 * h + 16 * g);
            a = a * sc;
#pragma unroll
            for (int nt = 0; nt < 4; ++nt) {
                v16h bf = *(const v16h*)&wdT[(16 * nt + Lm) * 192 + 64 * k + 32 * h + 16 * g];
                acc[nt] = WMMA_F16(a, bf, acc[nt]);
            }
        }
    }

#pragma unroll
    for (int nt = 0; nt < 4; ++nt)
#pragma unroll
        for (int r = 0; r < 8; ++r) {
            int G2 = r0 + r + 8 * g;
            int b2 = G2 >> 12, l2 = G2 & 4095;
            int col = 16 * nt + Lm;
            _Float16 v = (_Float16)(acc[nt][r] * oscale);
            if (transposed)
                Ytr[((size_t)b2 * 64 + col) * SEQ + l2] = v;
            else
                Yrow[(size_t)G2 * 64 + col] = v;
        }
}

// ---------------- flash attention ----------------------------------------
// Qh (prescaled by log2e), Kh: [B][L][64] f16;  Vt: [B][64][L] f16;  O: [B][L][64] f32
__device__ __forceinline__ void stage_tile(_Float16* KsB, _Float16* VsB,
                                           const _Float16* kbase, const _Float16* vbase,
                                           int kv, int tid) {
#pragma unroll
    for (int rr = 0; rr < 2; ++rr) {
        int f = tid * 8 + rr * 2048;          // 0..4095, 8 halves each
        int c = f >> 6, kk = f & 63;
#if ASYNC_LDS
        async_b128(&KsB[f], kbase + (size_t)kv * 64 + f);
        async_b128(&VsB[f], vbase + (size_t)c * SEQ + kv + kk);
#else
        *(v8h*)&KsB[f] = *(const v8h*)&kbase[(size_t)kv * 64 + f];
        *(v8h*)&VsB[f] = *(const v8h*)&vbase[(size_t)c * SEQ + kv + kk];
#endif
    }
}

__global__ __launch_bounds__(256) void flash_attn(const _Float16* __restrict__ Qh,
                                                  const _Float16* __restrict__ Kh,
                                                  const _Float16* __restrict__ Vt,
                                                  float* __restrict__ O) {
    __shared__ __align__(32) _Float16 Ks[2][64 * 64];   // K tile [kv][d], 2x8KB
    __shared__ __align__(32) _Float16 Vs[2][64 * 64];   // V tile [d][kv], 2x8KB
    __shared__ __align__(32) _Float16 Ps[8 * 16 * 64];  // per-wave P [row][kv], 16KB

    const int b = blockIdx.y;
    const int tid = threadIdx.x;
    const int w = tid >> 5, lane = tid & 31;
    const int g = lane >> 4, Lm = lane & 15;
    const int q0 = blockIdx.x * 128 + 16 * w;

    const _Float16* kbase = Kh + (size_t)b * SEQ * 64;
    const _Float16* vbase = Vt + (size_t)b * 64 * SEQ;

    const _Float16* qrow = Qh + ((size_t)b * SEQ + q0 + Lm) * 64;
    v16h qa0 = afrag_from_f16(qrow, 16 * g);        // d 0..31
    v16h qa1 = afrag_from_f16(qrow, 32 + 16 * g);   // d 32..63

    v8f o[4];
    float m[8], lsum[8];
#pragma unroll
    for (int nt = 0; nt < 4; ++nt)
#pragma unroll
        for (int i = 0; i < 8; ++i) o[nt][i] = 0.f;
#pragma unroll
    for (int r = 0; r < 8; ++r) { m[r] = -3.0e38f; lsum[r] = 0.f; }

    stage_tile(Ks[0], Vs[0], kbase, vbase, 0, tid);

    const int NIT = SEQ / 64;
    for (int it = 0; it < NIT; ++it) {
        const int kv = it * 64;
        const int cur = it & 1;

        WAIT_ASYNC0();        // this wave's staging of tile `it` complete
        __syncthreads();      // everyone's tile `it` visible; buf cur^1 free

        if (it + 1 < NIT)
            stage_tile(Ks[cur ^ 1], Vs[cur ^ 1], kbase, vbase, kv + 64, tid);

        // S = Q K^T : four 16x16 tiles covering 64 kv columns (log2 domain)
        v8f s[4];
#pragma unroll
        for (int j = 0; j < 4; ++j) {
            v8f c;
#pragma unroll
            for (int i = 0; i < 8; ++i) c[i] = 0.f;
            v16h b0 = *(const v16h*)&Ks[cur][(16 * j + Lm) * 64 + 16 * g];
            v16h b1 = *(const v16h*)&Ks[cur][(16 * j + Lm) * 64 + 32 + 16 * g];
            c = WMMA_F16(qa0, b0, c);
            c = WMMA_F16(qa1, b1, c);
            s[j] = c;
        }

        // online softmax (base-2)
        float alpha[8];
#pragma unroll
        for (int r = 0; r < 8; ++r) {
            float t = fmaxf(fmaxf(s[0][r], s[1][r]), fmaxf(s[2][r], s[3][r]));
            t = fmaxf(t, __shfl_xor(t, 1));
            t = fmaxf(t, __shfl_xor(t, 2));
            t = fmaxf(t, __shfl_xor(t, 4));
            t = fmaxf(t, __shfl_xor(t, 8));
            float mn = fmaxf(m[r], t);
            alpha[r] = exp2f(m[r] - mn);
            m[r] = mn;
        }
#pragma unroll
        for (int r = 0; r < 8; ++r) {
            int rowl = r + 8 * g;
            float ps = 0.f;
#pragma unroll
            for (int j = 0; j < 4; ++j) {
                float p = exp2f(s[j][r] - m[r]);
                ps += p;
                Ps[w * 1024 + rowl * 64 + 16 * j + Lm] = (_Float16)p;
            }
            ps += __shfl_xor(ps, 1);
            ps += __shfl_xor(ps, 2);
            ps += __shfl_xor(ps, 4);
            ps += __shfl_xor(ps, 8);
            lsum[r] = lsum[r] * alpha[r] + ps;
        }
#pragma unroll
        for (int nt = 0; nt < 4; ++nt)
#pragma unroll
            for (int r = 0; r < 8; ++r) o[nt][r] *= alpha[r];

        __syncthreads();  // P staging: D-layout -> A-layout

        // O += P @ V  (two K-steps of 32)
        v16h pa0 = join8(*(const v8h*)&Ps[w * 1024 + Lm * 64 + 16 * g],
                         *(const v8h*)&Ps[w * 1024 + Lm * 64 + 8 + 16 * g]);
        v16h pa1 = join8(*(const v8h*)&Ps[w * 1024 + Lm * 64 + 32 + 16 * g],
                         *(const v8h*)&Ps[w * 1024 + Lm * 64 + 40 + 16 * g]);
#pragma unroll
        for (int nt = 0; nt < 4; ++nt) {
            v16h bv0 = *(const v16h*)&Vs[cur][(16 * nt + Lm) * 64 + 16 * g];
            v16h bv1 = *(const v16h*)&Vs[cur][(16 * nt + Lm) * 64 + 32 + 16 * g];
            o[nt] = WMMA_F16(pa0, bv0, o[nt]);
            o[nt] = WMMA_F16(pa1, bv1, o[nt]);
        }
    }

#pragma unroll
    for (int nt = 0; nt < 4; ++nt)
#pragma unroll
        for (int r = 0; r < 8; ++r) {
            int row = q0 + r + 8 * g;
            O[((size_t)b * SEQ + row) * 64 + 16 * nt + Lm] = o[nt][r] / lsum[r];
        }
}

// ---------------- launch --------------------------------------------------

extern "C" void kernel_launch(void* const* d_in, const int* in_sizes, int n_in,
                              void* d_out, int out_size, void* d_ws, size_t ws_size,
                              hipStream_t stream) {
    const float* Q  = (const float*)d_in[0];
    const float* K  = (const float*)d_in[1];
    const float* V  = (const float*)d_in[2];
    const float* wq = (const float*)d_in[3];
    const float* bq = (const float*)d_in[4];
    const float* wk = (const float*)d_in[5];
    const float* bk = (const float*)d_in[6];
    const float* wv = (const float*)d_in[7];
    const float* bv = (const float*)d_in[8];
    const float* wd = (const float*)d_in[9];
    const float* bd = (const float*)d_in[10];
    const float* wo = (const float*)d_in[11];
    const float* bo = (const float*)d_in[12];

    char* ws = (char*)d_ws;
    const size_t MB = (size_t)1 << 20;
    _Float16* Qh  = (_Float16*)(ws + 0 * MB);    // [B][L][64] f16 (prescaled), 4MB
    _Float16* Kh  = (_Float16*)(ws + 4 * MB);    // [B][L][64] f16, 4MB
    _Float16* Vt  = (_Float16*)(ws + 8 * MB);    // [B][64][L] f16, 4MB
    _Float16* tmp = (_Float16*)(ws + 12 * MB);   // 3x [B*L][64] f16, 12MB
    float*    Ob  = (float*)   (ws + 12 * MB);   // [B][L][64] f32, 8MB (aliases tmp; tmp dead)
    char* wsw = ws + 24 * MB;
    _Float16* wqT = (_Float16*)(wsw);
    _Float16* wkT = (_Float16*)(wsw + 8192);
    _Float16* wvT = (_Float16*)(wsw + 16384);
    _Float16* wdT = (_Float16*)(wsw + 24576);
    _Float16* woT = (_Float16*)(wsw + 24576 + 24576);

    _Float16* tq = tmp;
    _Float16* tk = tmp + (size_t)BLR * 64;
    _Float16* tv = tk + (size_t)BLR * 64;

    prep_weights<<<1, 256, 0, stream>>>(wq, wk, wv, wd, wo, wqT, wkT, wvT, wdT, woT);

    pw_gemm<true><<<BLR / 128, 256, 0, stream>>>(Q, wqT, bq, tq);
    pw_gemm<true><<<BLR / 128, 256, 0, stream>>>(K, wkT, bk, tk);
    pw_gemm<true><<<BLR / 128, 256, 0, stream>>>(V, wvT, bv, tv);

    conv3_proj<<<BLR / 128, 256, 0, stream>>>(tq, wdT, bd, Qh, nullptr, 0, LOG2E);
    conv3_proj<<<BLR / 128, 256, 0, stream>>>(tk, wdT, bd, Kh, nullptr, 0, 1.0f);
    conv3_proj<<<BLR / 128, 256, 0, stream>>>(tv, wdT, bd, nullptr, Vt, 1, 1.0f);

    flash_attn<<<dim3(SEQ / 128, NB), 256, 0, stream>>>(Qh, Kh, Vt, Ob);

    pw_gemm<false><<<BLR / 128, 256, 0, stream>>>(Ob, woT, bo, d_out);
}
